// GCN_33122787787017
// MI455X (gfx1250) — matrix operand order repaired and checked by reading, not compiled, not run on hardware
//
#include <hip/hip_runtime.h>
#include <cstdint>
#include <cstddef>

// GCN dims (fixed by the reference)
#define IN_DIM 128
#define HID    32
#define OUT_DIM 20

typedef __attribute__((ext_vector_type(16))) __bf16 v16bf;
typedef __attribute__((ext_vector_type(8)))  float  v8f;

// ---------------------------------------------------------------------------
// Weight prep: Wt[n*K + k] = bf16(W[k*HID + n])  (transpose + convert, once)
// Makes each WMMA B-fragment 16 contiguous bf16 (32B -> 2x global_load_b128).
// ---------------------------------------------------------------------------
template <int K>
__global__ void k_prep_weights(const float* __restrict__ W, uint16_t* __restrict__ Wt) {
  int t = blockIdx.x * blockDim.x + threadIdx.x;
  if (t >= 32 * K) return;
  int n = t / K;
  int k = t - n * K;
  __bf16 v = (__bf16)W[(size_t)k * HID + n];
  Wt[(size_t)n * K + k] = __builtin_bit_cast(uint16_t, v);
}

// ---------------------------------------------------------------------------
// WMMA A fragment (16-bit 16x32; contiguous k order used here, same as B)
// lane L: g=L/16, m=L%16 holds row m:
//   elem j (0..15) -> K = g*16 + j
// f32 source, convert on the fly (pk-cvt co-executes with the WMMA pipe).
// ---------------------------------------------------------------------------
__device__ __forceinline__ v16bf load_a_frag(const float* __restrict__ p, int g) {
  const float4* q0 = reinterpret_cast<const float4*>(p + g * 16);
  const float4* q1 = reinterpret_cast<const float4*>(p + g * 16 + 8);
  float4 a0 = q0[0], a1 = q0[1];
  float4 c0 = q1[0], c1 = q1[1];
  v16bf a;
  a[0]=(__bf16)a0.x;  a[1]=(__bf16)a0.y;  a[2]=(__bf16)a0.z;  a[3]=(__bf16)a0.w;
  a[4]=(__bf16)a1.x;  a[5]=(__bf16)a1.y;  a[6]=(__bf16)a1.z;  a[7]=(__bf16)a1.w;
  a[8]=(__bf16)c0.x;  a[9]=(__bf16)c0.y;  a[10]=(__bf16)c0.z; a[11]=(__bf16)c0.w;
  a[12]=(__bf16)c1.x; a[13]=(__bf16)c1.y; a[14]=(__bf16)c1.z; a[15]=(__bf16)c1.w;
  return a;
}

// B: 32x16 (KxN) bf16 from pre-transposed Wt. lane L: g=L/16, n=L%16 holds
// column n, elem j -> K = k0 + g*16 + j  => 16 contiguous bf16 per lane.
template <int K>
__device__ __forceinline__ v16bf load_b_frag(const uint16_t* __restrict__ Wt,
                                             int k0, int n, int g) {
  return *reinterpret_cast<const v16bf*>(Wt + (size_t)n * K + k0 + g * 16);
}

// ---------------------------------------------------------------------------
// Y[nRows x 32] = X[nRows x K] @ W[K x 32]  (W pre-transposed bf16 in Wt).
// One wave per 16-row tile; two 16x16 N-tiles per wave. nRows % 16 == 0.
// ---------------------------------------------------------------------------
template <int K>
__global__ void gcn_gemm_wmma(const float* __restrict__ X,
                              const uint16_t* __restrict__ Wt,
                              float* __restrict__ Y, int nRows) {
  const int lane = threadIdx.x & 31;
  const int wave = blockIdx.x * (blockDim.x >> 5) + (threadIdx.x >> 5);
  const int row0 = wave * 16;
  if (row0 >= nRows) return;          // wave-uniform: EXEC stays all-ones for WMMA
  const int g = lane >> 4;            // half-wave group
  const int m = lane & 15;            // row within tile / column within N-tile
  const int row = row0 + m;

  v8f acc0 = {};                      // columns 0..15
  v8f acc1 = {};                      // columns 16..31
#pragma unroll
  for (int k0 = 0; k0 < K; k0 += 32) {
    v16bf a  = load_a_frag(X + (size_t)row * K + k0, g);
    v16bf b0 = load_b_frag<K>(Wt, k0, m, g);
    v16bf b1 = load_b_frag<K>(Wt, k0, 16 + m, g);
    acc0 = __builtin_amdgcn_wmma_f32_16x16x32_bf16(false, a, false, b0,
                                                   (short)0, acc0, false, false);
    acc1 = __builtin_amdgcn_wmma_f32_16x16x32_bf16(false, a, false, b1,
                                                   (short)0, acc1, false, false);
  }
  // D layout: VGPR r -> row (g*8 + r), lane -> column m (acc0) / 16+m (acc1)
  float* yb = Y + (size_t)(row0 + g * 8) * HID + m;
#pragma unroll
  for (int r = 0; r < 8; ++r) {
    yb[(size_t)r * HID]      = acc0[r];
    yb[(size_t)r * HID + 16] = acc1[r];
  }
}

// ---------------------------------------------------------------------------
// Small utility / edge kernels
// ---------------------------------------------------------------------------
__global__ void k_zero(float* __restrict__ p, size_t n) {
  size_t i = (size_t)blockIdx.x * blockDim.x + threadIdx.x;
  if (i < n) p[i] = 0.0f;
}

__global__ void k_degree(const int* __restrict__ col, float* __restrict__ deg, int E) {
  int e = blockIdx.x * blockDim.x + threadIdx.x;
  if (e < E) atomicAdd(&deg[col[e]], 1.0f);
}

__global__ void k_dis(const float* __restrict__ deg, float* __restrict__ dis, int n) {
  int i = blockIdx.x * blockDim.x + threadIdx.x;
  if (i < n) {
    float d = deg[i];
    dis[i] = (d > 0.0f) ? rsqrtf(d) : 0.0f;
  }
}

__global__ void k_norm(const int* __restrict__ row, const int* __restrict__ col,
                       const float* __restrict__ dis, float* __restrict__ nrm, int E) {
  int e = blockIdx.x * blockDim.x + threadIdx.x;
  if (e < E) nrm[e] = dis[row[e]] * dis[col[e]];
}

// agg[col[e]] += hW[row[e]] * norm[e]; 8 threads per edge, float4 per thread.
__global__ void k_scatter(const int* __restrict__ row, const int* __restrict__ col,
                          const float* __restrict__ nrm, const float* __restrict__ hW,
                          float* __restrict__ agg, int E) {
  size_t t = (size_t)blockIdx.x * blockDim.x + threadIdx.x;
  size_t e = t >> 3;
  int q = (int)(t & 7);
  if (e >= (size_t)E) return;
  int r = row[e];
  int c = col[e];
  float nm = nrm[e];
  const float4 v = *reinterpret_cast<const float4*>(hW + (size_t)r * HID + q * 4);
  float* dst = agg + (size_t)c * HID + q * 4;
  atomicAdd(dst + 0, v.x * nm);
  atomicAdd(dst + 1, v.y * nm);
  atomicAdd(dst + 2, v.z * nm);
  atomicAdd(dst + 3, v.w * nm);
}

// dst = tanh(src + b)  over [nNodes x HID]
__global__ void k_bias_tanh(const float* __restrict__ src, const float* __restrict__ b,
                            float* __restrict__ dst, int nNodes) {
  size_t i = (size_t)blockIdx.x * blockDim.x + threadIdx.x;
  if (i < (size_t)nNodes * HID) {
    int f = (int)(i & (HID - 1));
    dst[i] = tanhf(src[i] + b[f]);
  }
}

// out = log_softmax(h @ Wc + bc); one thread per node (HID=32 -> OUT=20)
__global__ void k_classifier(const float* __restrict__ h, const float* __restrict__ Wc,
                             const float* __restrict__ bc, float* __restrict__ out, int n) {
  int i = blockIdx.x * blockDim.x + threadIdx.x;
  if (i >= n) return;
  float hv[HID];
  const float4* hp = reinterpret_cast<const float4*>(h + (size_t)i * HID);
#pragma unroll
  for (int q = 0; q < HID / 4; ++q) {
    float4 v = hp[q];
    hv[4 * q + 0] = v.x; hv[4 * q + 1] = v.y;
    hv[4 * q + 2] = v.z; hv[4 * q + 3] = v.w;
  }
  float logits[OUT_DIM];
  float mx = -3.4e38f;
#pragma unroll
  for (int j = 0; j < OUT_DIM; ++j) {
    float s = bc[j];
#pragma unroll
    for (int f = 0; f < HID; ++f) s = fmaf(hv[f], Wc[f * OUT_DIM + j], s);
    logits[j] = s;
    mx = fmaxf(mx, s);
  }
  float se = 0.0f;
#pragma unroll
  for (int j = 0; j < OUT_DIM; ++j) se += expf(logits[j] - mx);
  float lse = mx + logf(se);
  float* op = out + (size_t)i * OUT_DIM;
#pragma unroll
  for (int j = 0; j < OUT_DIM; ++j) op[j] = logits[j] - lse;
}

// ---------------------------------------------------------------------------
// Host-side orchestration
// ---------------------------------------------------------------------------
extern "C" void kernel_launch(void* const* d_in, const int* in_sizes, int n_in,
                              void* d_out, int out_size, void* d_ws, size_t ws_size,
                              hipStream_t stream) {
  (void)n_in; (void)out_size; (void)ws_size;

  const float* x    = (const float*)d_in[0];
  const int*   ei   = (const int*)d_in[1];
  const float* W1   = (const float*)d_in[2];
  const float* b1   = (const float*)d_in[3];
  const float* W2   = (const float*)d_in[4];
  const float* b2   = (const float*)d_in[5];
  const float* Wc   = (const float*)d_in[6];
  const float* bc   = (const float*)d_in[7];

  const int N = in_sizes[0] / IN_DIM;   // 100000
  const int E = in_sizes[1] / 2;        // 3200000
  const int* rowI = ei;                 // source nodes
  const int* colI = ei + E;             // target nodes

  // Workspace layout:
  //   Wt1 bf16[32*128] (8192B) | Wt2 bf16[32*32] (2048B)   (10240B, 256B-aligned end)
  //   deg f32[N] | dis f32[N] | nrm f32[E] | hW f32[N*32] | agg f32[N*32]
  uint8_t* base = (uint8_t*)d_ws;
  uint16_t* Wt1 = (uint16_t*)base;
  uint16_t* Wt2 = Wt1 + 32 * IN_DIM;
  float* deg = (float*)(base + 10240);
  float* dis = deg + N;
  float* nrm = dis + N;
  float* hW  = nrm + (size_t)E;
  float* agg = hW + (size_t)N * HID;

  float* outLog = (float*)d_out;                 // [N, 20]
  float* hOut   = outLog + (size_t)N * OUT_DIM;  // [N, 32] (2nd tuple output)

  const int  T = 256;
  const unsigned gN    = (unsigned)((N + T - 1) / T);
  const unsigned gE    = (unsigned)((E + T - 1) / T);
  const unsigned gNH   = (unsigned)(((size_t)N * HID + T - 1) / T);
  const unsigned gScat = (unsigned)(((size_t)E * 8 + T - 1) / T);
  const unsigned gGemm = (unsigned)((N / 16 + 3) / 4);   // 4 waves (128 thr) per block

  // --- weight prep (bf16 transposed copies; tiny, L2-resident) ---
  k_prep_weights<IN_DIM><<<(32 * IN_DIM + T - 1) / T, T, 0, stream>>>(W1, Wt1);
  k_prep_weights<HID><<<(32 * HID + T - 1) / T, T, 0, stream>>>(W2, Wt2);

  // --- normalization ---
  k_zero<<<gN, T, 0, stream>>>(deg, (size_t)N);
  k_degree<<<gE, T, 0, stream>>>(colI, deg, E);
  k_dis<<<gN, T, 0, stream>>>(deg, dis, N);
  k_norm<<<gE, T, 0, stream>>>(rowI, colI, dis, nrm, E);

  // --- layer 1: hW = x @ W1 ; agg = scatter(norm * hW[row] -> col) ; h1 = tanh(agg+b1)
  k_zero<<<gNH, T, 0, stream>>>(agg, (size_t)N * HID);
  gcn_gemm_wmma<IN_DIM><<<gGemm, 128, 0, stream>>>(x, Wt1, hW, N);
  k_scatter<<<gScat, T, 0, stream>>>(rowI, colI, nrm, hW, agg, E);
  k_bias_tanh<<<gNH, T, 0, stream>>>(agg, b1, agg, N);   // h1 in-place in agg

  // --- layer 2: hW = h1 @ W2 ; re-zero agg ; scatter ; h2 = tanh(agg+b2) -> d_out
  gcn_gemm_wmma<HID><<<gGemm, 128, 0, stream>>>(agg, Wt2, hW, N);
  k_zero<<<gNH, T, 0, stream>>>(agg, (size_t)N * HID);
  k_scatter<<<gScat, T, 0, stream>>>(rowI, colI, nrm, hW, agg, E);
  k_bias_tanh<<<gNH, T, 0, stream>>>(agg, b2, hOut, N);  // tuple output h

  // --- classifier head: log_softmax(h2 @ Wc + bc) -> d_out[:, :20]
  k_classifier<<<gN, T, 0, stream>>>(hOut, Wc, bc, outLog, N);
}
